// DenseAttention_85074712199779
// MI455X (gfx1250) — matrix operand order repaired and checked by reading, not compiled, not run on hardware
//
#include <hip/hip_runtime.h>

#define BB 8
#define NN 8192
#define DD 512
#define HH 8
#define DHD 64
#define SPLITS 16
#define CHUNK (NN / SPLITS) // 512

typedef __attribute__((ext_vector_type(16))) __bf16 v16bf;
typedef __attribute__((ext_vector_type(8)))  float  v8f;
typedef __attribute__((ext_vector_type(4)))  int    i32x4;

struct Frag { union { v16bf v; uint4 q[2]; }; };

#if defined(__has_builtin)
#if __has_builtin(__builtin_amdgcn_global_load_async_to_lds_b128)
#define USE_ASYNC_LDS 1
#endif
#endif

// Builtin param types (from hipcc diagnostic): (int4 AS1* src, int4 AS3* dst, Ii, Ii)
typedef __attribute__((address_space(1))) i32x4* gvec_ptr;
typedef __attribute__((address_space(3))) i32x4* lvec_ptr;

__device__ inline void wait_async0() {
#if __has_builtin(__builtin_amdgcn_s_wait_asynccnt)
  __builtin_amdgcn_s_wait_asynccnt(0);
#else
  asm volatile("s_wait_asynccnt 0x0" ::: "memory");
#endif
}

__device__ inline unsigned short f2bf(float f) {
  union { float f; unsigned u; } c; c.f = f;
  unsigned r = ((c.u >> 16) & 1u) + 0x7FFFu;   // round-to-nearest-even
  return (unsigned short)((c.u + r) >> 16);
}

// ---------------------------------------------------------------------------
// 1) RoPE: x = h*cos + rotate_half(h)*sin, emitted as bf16.
// ---------------------------------------------------------------------------
__global__ __launch_bounds__(256) void rope_kernel(
    const float* __restrict__ hs, const float* __restrict__ cosb,
    const float* __restrict__ sinb, unsigned short* __restrict__ xbf) {
  int t   = blockIdx.x * 256 + threadIdx.x;  // 4,194,304 threads
  int row = t >> 6;                          // b*N + n   (65536 rows)
  int d0  = (t & 63) << 2;                   // 0..252 step 4
  int n   = row & (NN - 1);
  const float4 h1 = *(const float4*)(hs   + (size_t)row * DD + d0);
  const float4 h2 = *(const float4*)(hs   + (size_t)row * DD + 256 + d0);
  const float4 c1 = *(const float4*)(cosb + (size_t)n   * DD + d0);
  const float4 s1 = *(const float4*)(sinb + (size_t)n   * DD + d0);
  const float4 c2 = *(const float4*)(cosb + (size_t)n   * DD + 256 + d0);
  const float4 s2 = *(const float4*)(sinb + (size_t)n   * DD + 256 + d0);
  float4 o1, o2;
  o1.x = h1.x * c1.x - h2.x * s1.x;  o2.x = h2.x * c2.x + h1.x * s2.x;
  o1.y = h1.y * c1.y - h2.y * s1.y;  o2.y = h2.y * c2.y + h1.y * s2.y;
  o1.z = h1.z * c1.z - h2.z * s1.z;  o2.z = h2.z * c2.z + h1.z * s2.z;
  o1.w = h1.w * c1.w - h2.w * s1.w;  o2.w = h2.w * c2.w + h1.w * s2.w;
  uint2 p1, p2;
  p1.x = (unsigned)f2bf(o1.x) | ((unsigned)f2bf(o1.y) << 16);
  p1.y = (unsigned)f2bf(o1.z) | ((unsigned)f2bf(o1.w) << 16);
  p2.x = (unsigned)f2bf(o2.x) | ((unsigned)f2bf(o2.y) << 16);
  p2.y = (unsigned)f2bf(o2.z) | ((unsigned)f2bf(o2.w) << 16);
  *(uint2*)(xbf + (size_t)row * DD + d0)       = p1;
  *(uint2*)(xbf + (size_t)row * DD + 256 + d0) = p2;
}

// ---------------------------------------------------------------------------
// 2) kv partial: per (split s, b, h) compute 64x64 partial of X_h^T X_h.
// Double-buffered LDS, software-pipelined global loads, packed b32 transpose.
// ---------------------------------------------------------------------------
__global__ __launch_bounds__(128) void kv_partial_kernel(
    const unsigned short* __restrict__ xbf, float* __restrict__ part) {
  __shared__ unsigned short xT[2][64][40]; // [buf][d][n_local]
  int s  = blockIdx.x & (SPLITS - 1);
  int bh = blockIdx.x >> 4;        // b*8 + h
  int hh = bh & 7, b = bh >> 3;
  int tid = threadIdx.x, lane = tid & 31, w = tid >> 5;
  int l = lane & 15, hi = lane >> 4;

  v8f acc[4] = {};
  const unsigned short* xb = xbf + (size_t)b * NN * DD + (size_t)hh * DHD;
  int n0 = s * CHUNK;

  // staging task: an 8d x 2n block per thread (128 tasks == 128 threads)
  int k2 = tid >> 3;          // n-pair index 0..15
  int d0 = (tid & 7) << 3;    // 0..56

  uint4 r0, r1;
  auto loadRegs = [&](int ktAbs) {
    const unsigned short* p = xb + (size_t)(n0 + ktAbs + 2 * k2) * DD + d0;
    r0 = *(const uint4*)p;
    r1 = *(const uint4*)(p + DD);
  };
  auto storeTile = [&](int buf) {
    union { uint4 q; unsigned short u[8]; } a0, a1;
    a0.q = r0; a1.q = r1;
    #pragma unroll
    for (int j = 0; j < 8; ++j) {
      unsigned pr = (unsigned)a0.u[j] | ((unsigned)a1.u[j] << 16);
      *(unsigned*)&xT[buf][d0 + j][2 * k2] = pr;   // ds_store_b32
    }
  };

  loadRegs(0);
  storeTile(0);
  __syncthreads();

  const int NK = CHUNK / 32;  // 16
  for (int kt = 0; kt < NK; ++kt) {
    if (kt + 1 < NK) loadRegs((kt + 1) * 32);
    int buf = kt & 1;
    Frag a, bf[4];
    {
      int r = 16 * w + l, klo = hi ? 8 : 0;
      a.q[0] = *(const uint4*)&xT[buf][r][klo];
      a.q[1] = *(const uint4*)&xT[buf][r][klo + 16];
    }
    #pragma unroll
    for (int j = 0; j < 4; ++j) {
      int r = 16 * j + l, kb = hi ? 8 : 0;
      bf[j].q[0] = *(const uint4*)&xT[buf][r][kb];
      bf[j].q[1] = *(const uint4*)&xT[buf][r][kb + 16];
    }
    #pragma unroll
    for (int j = 0; j < 4; ++j)
      acc[j] = __builtin_amdgcn_wmma_f32_16x16x32_bf16(
          false, a.v, false, bf[j].v, (short)0, acc[j], false, false);
    if (kt + 1 < NK) storeTile((kt + 1) & 1);
    __syncthreads();
  }

  float* pb = part + ((size_t)s * 64 + bh) * 4096;
  #pragma unroll
  for (int j = 0; j < 4; ++j) {
    int e = 16 * j + l;
    #pragma unroll
    for (int r = 0; r < 8; ++r) {
      int d = 16 * w + (hi ? 8 : 0) + r;
      pb[d * DHD + e] = acc[j][r];
    }
  }
}

// ---------------------------------------------------------------------------
// 3) Reduce 16 split-K partials -> bf16 kv (symmetric, reused transposed).
// ---------------------------------------------------------------------------
__global__ __launch_bounds__(256) void kv_reduce_kernel(
    const float* __restrict__ part, unsigned short* __restrict__ kvbf) {
  int g = blockIdx.x * 256 + threadIdx.x;  // 262144 elems
  float s = 0.f;
  #pragma unroll
  for (int i = 0; i < SPLITS; ++i) s += part[(size_t)i * 262144 + g];
  kvbf[g] = f2bf(s);
}

// ---------------------------------------------------------------------------
// 4) WeffT[b][f][c] = sum_d W[h*64+d][c] * kv[b,h][d][e]  (f = h*64+e),
// stored N-major so GEMM B-fragments read contiguous K.
// ---------------------------------------------------------------------------
__global__ __launch_bounds__(128) void weff_kernel(
    const float* __restrict__ W, const unsigned short* __restrict__ kvbf,
    unsigned short* __restrict__ weffT) {
  __shared__ unsigned short aT[64][40];  // [c_local][d]
  int c0t = blockIdx.x & 7;
  int hh  = (blockIdx.x >> 3) & 7;
  int b   = blockIdx.x >> 6;
  int c0  = c0t * 64;
  int tid = threadIdx.x, lane = tid & 31, w = tid >> 5;
  int l = lane & 15, hi = lane >> 4;

  #pragma unroll
  for (int pass = 0; pass < 8; ++pass) {
    int t  = tid + pass * 128;  // 1024 tasks: 64 d-rows x 16 float4
    int d  = t >> 4;
    int c4 = (t & 15) << 2;
    float4 wv = *(const float4*)(W + (size_t)(hh * DHD + d) * DD + c0 + c4);
    aT[c4 + 0][d] = f2bf(wv.x);
    aT[c4 + 1][d] = f2bf(wv.y);
    aT[c4 + 2][d] = f2bf(wv.z);
    aT[c4 + 3][d] = f2bf(wv.w);
  }
  __syncthreads();

  const unsigned short* kvb = kvbf + ((size_t)b * HH + hh) * 4096;
  v8f acc[4] = {};
  #pragma unroll
  for (int ks = 0; ks < 2; ++ks) {
    int kd0 = ks * 32;
    Frag a, bf[4];
    int r = 16 * w + l;
    int klo = (hi ? 8 : 0) + kd0;
    a.q[0] = *(const uint4*)&aT[r][klo];
    a.q[1] = *(const uint4*)&aT[r][klo + 16];
    #pragma unroll
    for (int j = 0; j < 4; ++j) {
      int e  = 16 * j + l;
      int kb = (hi ? 8 : 0) + kd0;
      bf[j].q[0] = *(const uint4*)(kvb + e * DHD + kb);      // kv symmetric
      bf[j].q[1] = *(const uint4*)(kvb + e * DHD + kb + 16);
    }
    #pragma unroll
    for (int j = 0; j < 4; ++j)
      acc[j] = __builtin_amdgcn_wmma_f32_16x16x32_bf16(
          false, a.v, false, bf[j].v, (short)0, acc[j], false, false);
  }
  #pragma unroll
  for (int j = 0; j < 4; ++j) {
    int f  = hh * DHD + 16 * j + l;
    int cb = c0 + 16 * w + (hi ? 8 : 0);
    union { uint4 q; unsigned short u[8]; } v;
    #pragma unroll
    for (int r = 0; r < 8; ++r) v.u[r] = f2bf(acc[j][r]);
    *(uint4*)(weffT + ((size_t)b * DD + f) * DD + cb) = v.q;
  }
}

// ---------------------------------------------------------------------------
// 5) Main GEMM: out[b] = x_bf[b] (8192x512) * WeffT[b]^T (512x512), f32 out.
// 128x128 WG tile, 8 waves (4M x 2N). Double-buffered LDS; staging uses
// GLOBAL_LOAD_ASYNC_TO_LDS_B128 (ASYNCcnt) when available.
// ---------------------------------------------------------------------------
__global__ __launch_bounds__(256) void gemm_kernel(
    const unsigned short* __restrict__ xbf,
    const unsigned short* __restrict__ weffT, float* __restrict__ out) {
  __shared__ unsigned short As[2][128][40];
  __shared__ unsigned short Bs[2][128][40];
  int mt = blockIdx.x & 63;
  int nt = blockIdx.x >> 6;
  int b  = blockIdx.y;
  int n0 = mt * 128;
  int f0 = nt * 128;
  int tid = threadIdx.x, lane = tid & 31, w = tid >> 5;
  int l = lane & 15, hi = lane >> 4;
  int wm = w & 3, wn = w >> 2;

  const unsigned short* Abase = xbf   + ((size_t)b * NN + n0) * DD;
  const unsigned short* Bbase = weffT + ((size_t)b * DD + f0) * DD;

  // staging: 512 b128 chunks per matrix -> 2 A-chunks + 2 B-chunks per thread
  int t0 = tid, t1 = tid + 256;
  int rA0 = t0 >> 2, c0 = (t0 & 3) << 3;
  int rA1 = t1 >> 2, c1 = (t1 & 3) << 3;

  v8f acc[2][4] = {};

#ifdef USE_ASYNC_LDS
  auto issueTile = [&](int kk, int buf) {
    __builtin_amdgcn_global_load_async_to_lds_b128(
        (gvec_ptr)(Abase + (size_t)rA0 * DD + kk + c0), (lvec_ptr)&As[buf][rA0][c0], 0, 0);
    __builtin_amdgcn_global_load_async_to_lds_b128(
        (gvec_ptr)(Abase + (size_t)rA1 * DD + kk + c1), (lvec_ptr)&As[buf][rA1][c1], 0, 0);
    __builtin_amdgcn_global_load_async_to_lds_b128(
        (gvec_ptr)(Bbase + (size_t)rA0 * DD + kk + c0), (lvec_ptr)&Bs[buf][rA0][c0], 0, 0);
    __builtin_amdgcn_global_load_async_to_lds_b128(
        (gvec_ptr)(Bbase + (size_t)rA1 * DD + kk + c1), (lvec_ptr)&Bs[buf][rA1][c1], 0, 0);
  };
  issueTile(0, 0);
  wait_async0();
  __syncthreads();
#else
  uint4 ra0, ra1, rb0, rb1;
  auto loadRegs = [&](int kk) {
    ra0 = *(const uint4*)(Abase + (size_t)rA0 * DD + kk + c0);
    ra1 = *(const uint4*)(Abase + (size_t)rA1 * DD + kk + c1);
    rb0 = *(const uint4*)(Bbase + (size_t)rA0 * DD + kk + c0);
    rb1 = *(const uint4*)(Bbase + (size_t)rA1 * DD + kk + c1);
  };
  auto storeTile = [&](int buf) {
    *(uint4*)&As[buf][rA0][c0] = ra0;
    *(uint4*)&As[buf][rA1][c1] = ra1;
    *(uint4*)&Bs[buf][rA0][c0] = rb0;
    *(uint4*)&Bs[buf][rA1][c1] = rb1;
  };
  loadRegs(0);
  storeTile(0);
  __syncthreads();
#endif

  const int NK = DD / 32;  // 16
  for (int kt = 0; kt < NK; ++kt) {
    int buf = kt & 1;
#ifdef USE_ASYNC_LDS
    if (kt + 1 < NK) issueTile((kt + 1) * 32, (kt + 1) & 1);
#else
    if (kt + 1 < NK) loadRegs((kt + 1) * 32);
#endif
    Frag a[2], bf[4];
    #pragma unroll
    for (int ms = 0; ms < 2; ++ms) {
      int r = wm * 32 + ms * 16 + l;
      int klo = hi ? 8 : 0;
      a[ms].q[0] = *(const uint4*)&As[buf][r][klo];
      a[ms].q[1] = *(const uint4*)&As[buf][r][klo + 16];
    }
    #pragma unroll
    for (int j = 0; j < 4; ++j) {
      int r  = wn * 64 + 16 * j + l;
      int kb = hi ? 8 : 0;
      bf[j].q[0] = *(const uint4*)&Bs[buf][r][kb];
      bf[j].q[1] = *(const uint4*)&Bs[buf][r][kb + 16];
    }
    #pragma unroll
    for (int ms = 0; ms < 2; ++ms)
      #pragma unroll
      for (int j = 0; j < 4; ++j)
        acc[ms][j] = __builtin_amdgcn_wmma_f32_16x16x32_bf16(
            false, a[ms].v, false, bf[j].v, (short)0, acc[ms][j], false, false);
#ifdef USE_ASYNC_LDS
    if (kt + 1 < NK) wait_async0();
#else
    if (kt + 1 < NK) storeTile((kt + 1) & 1);
#endif
    __syncthreads();
  }

  #pragma unroll
  for (int ms = 0; ms < 2; ++ms) {
    #pragma unroll
    for (int j = 0; j < 4; ++j) {
      int f  = f0 + wn * 64 + 16 * j + l;
      int nb = n0 + wm * 32 + ms * 16 + (hi ? 8 : 0);
      #pragma unroll
      for (int r = 0; r < 8; ++r)
        out[((size_t)b * NN + nb + r) * DD + f] = acc[ms][j][r];
    }
  }
}

// ---------------------------------------------------------------------------
extern "C" void kernel_launch(void* const* d_in, const int* in_sizes, int n_in,
                              void* d_out, int out_size, void* d_ws, size_t ws_size,
                              hipStream_t stream) {
  (void)in_sizes; (void)n_in; (void)out_size; (void)ws_size;
  const float* hs   = (const float*)d_in[0];
  const float* W    = (const float*)d_in[1];
  const float* cosb = (const float*)d_in[2];
  const float* sinb = (const float*)d_in[3];
  float* out = (float*)d_out;

  char* ws = (char*)d_ws;
  unsigned short* xbf   = (unsigned short*)ws;                         // 64 MiB bf16 x
  float*          part  = (float*)(ws + 67108864);                     // 16 MiB kv partials
  unsigned short* kvbf  = (unsigned short*)(ws + 67108864 + 16777216); // 512 KiB kv bf16
  unsigned short* weffT = (unsigned short*)(ws + 67108864 + 16777216 + 524288); // 4 MiB

  rope_kernel<<<16384, 256, 0, stream>>>(hs, cosb, sinb, xbf);
  kv_partial_kernel<<<BB * HH * SPLITS, 128, 0, stream>>>(xbf, part);
  kv_reduce_kernel<<<1024, 256, 0, stream>>>(part, kvbf);
  weff_kernel<<<BB * HH * 8, 128, 0, stream>>>(W, kvbf, weffT);
  gemm_kernel<<<dim3(256, 8), 256, 0, stream>>>(xbf, weffT, out);
}
